// Attention_76072460747440
// MI455X (gfx1250) — matrix operand order repaired and checked by reading, not compiled, not run on hardware
//
#include <hip/hip_runtime.h>

// ---------------- Problem constants (from reference) ----------------
#define Bc 2
#define Tc 2048
#define Dc 2048
#define Nc 8
#define Kc 4
#define Hc 256
#define WINDOW 1024
#define SCALEQ 0.0625f

// ---------------- WMMA helpers (gfx1250, wave32) ----------------
typedef __attribute__((ext_vector_type(16))) __bf16 v16bf;
typedef __attribute__((ext_vector_type(8)))  float  v8f;

union ABFrag { v16bf v; unsigned short u[16]; };

// native f32 -> bf16 (clang emits v_cvt_pk_bf16_f32 for pairs)
__device__ __forceinline__ unsigned short f2bf(float f) {
  union { __bf16 b; unsigned short u; } cv;
  cv.b = (__bf16)f;
  return cv.u;
}

__device__ __forceinline__ v8f wmma_bf16(const ABFrag& a, const ABFrag& b, v8f c) {
  return __builtin_amdgcn_wmma_f32_16x16x32_bf16(false, a.v, false, b.v,
                                                 (short)0, c, false, false);
}

// =====================================================================
// Kernel 1/4: generic bf16-WMMA GEMM, 128x64 tile, 8 waves, k-step 32,
// double-buffered LDS staging (global loads overlap the WMMA stream;
// one barrier per k-iteration). All strides are compile-time constants
// so address math folds into immediate offsets.
// =====================================================================
#define BM 128
#define BN 64
#define BKK 32

template<int LDA, int LDB, int LDO, int KD>
__global__ __launch_bounds__(256) void gemm_wmma(
    const float* __restrict__ A,
    const float* __restrict__ Bbase, long bHeadStride,
    float* __restrict__ Obase, long oHeadOff)
{
  __shared__ unsigned short As [2][BM][BKK + 8];   // [buf][m][k]
  __shared__ unsigned short BsT[2][BN][BKK + 8];   // [buf][n][k] (B transposed)

  const int tid    = threadIdx.x;
  const int wave   = tid >> 5;
  const int lane   = tid & 31;
  const int lanelo = lane & 15;
  const int hi     = lane >> 4;
  const int m0     = blockIdx.y * BM;
  const int n0     = blockIdx.x * BN;
  const float* Bp  = Bbase + (long)blockIdx.z * bHeadStride;
  float*       Op  = Obase + (long)blockIdx.z * oHeadOff;

  float4 fa[4], fb[2];

  auto gload = [&](int k0) {
    #pragma unroll
    for (int i = 0; i < 4; ++i) {
      int idx = tid + i * 256;
      int r  = idx >> 3;        // row 0..127
      int c4 = idx & 7;
      const float* ap = A + (long)(m0 + r) * LDA + k0 + c4 * 4;
      fa[i] = *(const float4*)ap;
      __builtin_prefetch(ap + 2 * BKK, 0, 3);       // speculative, k0+2 tiles
    }
    #pragma unroll
    for (int i = 0; i < 2; ++i) {
      int idx = tid + i * 256;
      int r  = idx >> 4;        // k row 0..31
      int c4 = idx & 15;
      const float* bp = Bp + (long)(k0 + r) * LDB + n0 + c4 * 4;
      fb[i] = *(const float4*)bp;
      __builtin_prefetch(bp + (long)2 * BKK * LDB, 0, 3);
    }
  };
  auto lstore = [&](int buf) {
    #pragma unroll
    for (int i = 0; i < 4; ++i) {
      int idx = tid + i * 256;
      int r  = idx >> 3;
      int c4 = idx & 7;
      unsigned short* d = &As[buf][r][c4 * 4];
      d[0] = f2bf(fa[i].x); d[1] = f2bf(fa[i].y);
      d[2] = f2bf(fa[i].z); d[3] = f2bf(fa[i].w);
    }
    #pragma unroll
    for (int i = 0; i < 2; ++i) {
      int idx = tid + i * 256;
      int r  = idx >> 4;
      int c4 = idx & 15;
      BsT[buf][c4 * 4 + 0][r] = f2bf(fb[i].x);
      BsT[buf][c4 * 4 + 1][r] = f2bf(fb[i].y);
      BsT[buf][c4 * 4 + 2][r] = f2bf(fb[i].z);
      BsT[buf][c4 * 4 + 3][r] = f2bf(fb[i].w);
    }
  };

  v8f acc[4];
  #pragma unroll
  for (int j = 0; j < 4; ++j)
    #pragma unroll
    for (int e = 0; e < 8; ++e) acc[j][e] = 0.0f;

  // prologue: fill buffer 0
  gload(0);
  lstore(0);
  __syncthreads();

  for (int k0 = 0; k0 < KD; k0 += BKK) {
    const int  buf  = (k0 / BKK) & 1;
    const bool more = (k0 + BKK) < KD;
    if (more) gload(k0 + BKK);      // in flight while we compute

    // A fragment: two contiguous 8-short runs per lane -> 2x ds_load_b128
    ABFrag a;
    {
      const unsigned short* row = &As[buf][wave * 16 + lanelo][hi * 8];
      #pragma unroll
      for (int e = 0; e < 8; ++e) a.u[e] = row[e];
      #pragma unroll
      for (int e = 0; e < 8; ++e) a.u[8 + e] = row[16 + e];
    }
    #pragma unroll
    for (int j = 0; j < 4; ++j) {
      ABFrag b;   // b.u[e] = B[k=8*hi+e (+8 for e>=8)][n=j*16+lanelo] = BsT[n][k], two runs
      const unsigned short* row = &BsT[buf][j * 16 + lanelo][hi * 8];
      #pragma unroll
      for (int e = 0; e < 8; ++e) b.u[e] = row[e];
      #pragma unroll
      for (int e = 0; e < 8; ++e) b.u[8 + e] = row[16 + e];
      acc[j] = wmma_bf16(a, b, acc[j]);
    }

    if (more) lstore(buf ^ 1);      // convert + store into the other buffer
    __syncthreads();
  }

  #pragma unroll
  for (int j = 0; j < 4; ++j)
    #pragma unroll
    for (int r = 0; r < 8; ++r) {
      int mm = m0 + wave * 16 + r + hi * 8;
      int nn = n0 + j * 16 + lanelo;
      Op[(long)mm * LDO + nn] = acc[j][r];
    }
}

// =====================================================================
// Kernel 2: fused RMSNorm + RoPE, one wave per 256-element head row.
// =====================================================================
__global__ __launch_bounds__(256) void normrope(
    float* __restrict__ data, const float* __restrict__ scale,
    const int* __restrict__ segpos, int heads, int nrows)
{
  int row = blockIdx.x * 8 + (threadIdx.x >> 5);
  if (row >= nrows) return;
  int lane = threadIdx.x & 31;
  int hi   = lane >> 4;
  int pos  = segpos[row / heads];           // row = (b*T+t)*heads + n

  float* p = data + (long)row * Hc + lane * 8;
  float x[8];
  float4 v0 = *(const float4*)p;
  float4 v1 = *(const float4*)(p + 4);
  x[0]=v0.x; x[1]=v0.y; x[2]=v0.z; x[3]=v0.w;
  x[4]=v1.x; x[5]=v1.y; x[6]=v1.z; x[7]=v1.w;

  float ss = 0.f;
  #pragma unroll
  for (int e = 0; e < 8; ++e) ss += x[e] * x[e];
  #pragma unroll
  for (int off = 16; off >= 1; off >>= 1) ss += __shfl_xor(ss, off, 32);
  float rinv = rsqrtf(ss * (1.0f / (float)Hc) + 1e-6f);

  float y[8];
  #pragma unroll
  for (int e = 0; e < 8; ++e) {
    int h = lane * 8 + e;
    y[e] = x[e] * rinv * scale[h];
  }
  // RoPE: pair (h, h+128) lives in lane ^ 16 (exact half swap in wave32)
  float out[8];
  #pragma unroll
  for (int e = 0; e < 8; ++e) {
    float partner = __shfl_xor(y[e], 16, 32);
    int i = (lane * 8 + e) & 127;
    float inv_ts = __expf(-(2.0f * (float)i / (float)Hc) * 9.210340371976184f); // 1/10000^(2i/H)
    float ang = (float)pos * inv_ts;
    float s, c;
    __sincosf(ang, &s, &c);
    out[e] = (hi == 0) ? (y[e] * c - partner * s) : (y[e] * c + partner * s);
  }
  float4 w0 = {out[0], out[1], out[2], out[3]};
  float4 w1 = {out[4], out[5], out[6], out[7]};
  *(float4*)p = w0;
  *(float4*)(p + 4) = w1;
}

// =====================================================================
// Kernel 3: flash attention with sliding window, bf16 WMMA.
// Block = 128 threads (4 waves); block covers 64 q-rows of one (b, head);
// each wave owns 16 q rows. Key tiles of 32 staged in LDS.
// K stored [key][h] (QK^T B-frag contiguous in h); V stored transposed
// [h][key] (PV B-frag contiguous in key).
// =====================================================================
#define QT 64
#define KT 32

__global__ __launch_bounds__(128) void attn_wmma(
    const float* __restrict__ qws,   // [B,T,N,H]
    const float* __restrict__ kws,   // [B,T,K,H]
    const float* __restrict__ vws,   // [B,T,K,H]
    float* __restrict__ enc)         // [B,T,N,H]
{
  __shared__ unsigned short Ks [KT][Hc + 8];   // [key][h]
  __shared__ unsigned short VsT[Hc][KT + 8];   // [h][key]
  __shared__ unsigned short Ps [4][16][KT + 8];

  const int qt = blockIdx.x, n = blockIdx.y, b = blockIdx.z;
  const int kv = n / (Nc / Kc);
  const int tid = threadIdx.x, wave = tid >> 5, lane = tid & 31;
  const int lanelo = lane & 15, hi = lane >> 4;
  const int i0 = qt * QT;
  const int qrowBase = i0 + wave * 16;

  // Q resident in registers: 8 pre-scaled A fragments (16 rows x 256),
  // loaded as 4x float4 per fragment.
  ABFrag aq[8];
  {
    const float* qrow = qws + ((long)(b * Tc + qrowBase + lanelo) * Nc + n) * Hc;
    #pragma unroll
    for (int ks = 0; ks < 8; ++ks) {
      const float* base = qrow + ks * 32 + hi * 8;
      float4 f0 = *(const float4*)(base);
      float4 f1 = *(const float4*)(base + 4);
      float4 f2 = *(const float4*)(base + 16);
      float4 f3 = *(const float4*)(base + 20);
      unsigned short* u = aq[ks].u;
      u[0]=f2bf(f0.x*SCALEQ); u[1]=f2bf(f0.y*SCALEQ); u[2]=f2bf(f0.z*SCALEQ); u[3]=f2bf(f0.w*SCALEQ);
      u[4]=f2bf(f1.x*SCALEQ); u[5]=f2bf(f1.y*SCALEQ); u[6]=f2bf(f1.z*SCALEQ); u[7]=f2bf(f1.w*SCALEQ);
      u[8]=f2bf(f2.x*SCALEQ); u[9]=f2bf(f2.y*SCALEQ); u[10]=f2bf(f2.z*SCALEQ); u[11]=f2bf(f2.w*SCALEQ);
      u[12]=f2bf(f3.x*SCALEQ); u[13]=f2bf(f3.y*SCALEQ); u[14]=f2bf(f3.z*SCALEQ); u[15]=f2bf(f3.w*SCALEQ);
    }
  }

  v8f o[16];
  #pragma unroll
  for (int j = 0; j < 16; ++j)
    #pragma unroll
    for (int e = 0; e < 8; ++e) o[j][e] = 0.0f;
  float rmax[8], rsum[8];
  #pragma unroll
  for (int r = 0; r < 8; ++r) { rmax[r] = -1e30f; rsum[r] = 0.0f; }

  int tlo = i0 - (WINDOW - 1);
  if (tlo < 0) tlo = 0;
  tlo &= ~(KT - 1);
  const int thi = i0 + QT - 1;

  for (int s0 = tlo; s0 <= thi; s0 += KT) {
    // stage K [key][h] and V^T [h][key] tiles (f32 -> bf16)
    #pragma unroll
    for (int it = 0; it < 16; ++it) {
      int i = tid + it * 128;
      int r  = i >> 6;     // key row 0..31
      int c4 = i & 63;     // float4 col along h
      long base = ((long)(b * Tc + s0 + r) * Kc + kv) * Hc + c4 * 4;
      float4 f = *(const float4*)(kws + base);
      float4 g = *(const float4*)(vws + base);
      __builtin_prefetch(kws + base + (long)KT * Kc * Hc, 0, 3);  // speculative
      __builtin_prefetch(vws + base + (long)KT * Kc * Hc, 0, 3);
      unsigned short* dk = &Ks[r][c4 * 4];
      dk[0]=f2bf(f.x); dk[1]=f2bf(f.y); dk[2]=f2bf(f.z); dk[3]=f2bf(f.w);
      int h = c4 * 4;
      VsT[h + 0][r] = f2bf(g.x);
      VsT[h + 1][r] = f2bf(g.y);
      VsT[h + 2][r] = f2bf(g.z);
      VsT[h + 3][r] = f2bf(g.w);
    }
    __syncthreads();

    // S = Q * K^T  (16 x 32 per wave) in two 16x16 col tiles
    v8f s[2];
    #pragma unroll
    for (int j = 0; j < 2; ++j)
      #pragma unroll
      for (int e = 0; e < 8; ++e) s[j][e] = 0.0f;
    #pragma unroll
    for (int ks = 0; ks < 8; ++ks) {
      ABFrag bk[2];   // B[kdim=h][col=key]: Ks[key][h] contiguous in h
      #pragma unroll
      for (int j = 0; j < 2; ++j) {
        const unsigned short* row = &Ks[j * 16 + lanelo][ks * 32 + hi * 8];
        #pragma unroll
        for (int e = 0; e < 8; ++e) bk[j].u[e] = row[e];
        #pragma unroll
        for (int e = 0; e < 8; ++e) bk[j].u[8 + e] = row[16 + e];
      }
      #pragma unroll
      for (int j = 0; j < 2; ++j)
        s[j] = wmma_bf16(aq[ks], bk[j], s[j]);
    }

    // mask (causal AND window) + online softmax
    float alpha[8];
    #pragma unroll
    for (int r = 0; r < 8; ++r) {
      int qrow = qrowBase + r + hi * 8;
      float mx = -1e30f;
      #pragma unroll
      for (int j = 0; j < 2; ++j) {
        int key = s0 + j * 16 + lanelo;
        float val = s[j][r];
        bool ok = (key <= qrow) && (key >= qrow - (WINDOW - 1));
        val = ok ? val : -1e30f;
        s[j][r] = val;
        mx = fmaxf(mx, val);
      }
      #pragma unroll
      for (int off = 8; off >= 1; off >>= 1)
        mx = fmaxf(mx, __shfl_xor(mx, off, 32));   // within 16-lane half
      float mnew = fmaxf(rmax[r], mx);
      alpha[r] = __expf(rmax[r] - mnew);
      rmax[r] = mnew;
      float acc = 0.f;
      #pragma unroll
      for (int j = 0; j < 2; ++j) {
        float pr = __expf(s[j][r] - mnew);
        s[j][r] = pr;
        acc += pr;
      }
      #pragma unroll
      for (int off = 8; off >= 1; off >>= 1)
        acc += __shfl_xor(acc, off, 32);
      rsum[r] = rsum[r] * alpha[r] + acc;
    }
    // rescale O accumulators
    #pragma unroll
    for (int j = 0; j < 16; ++j)
      #pragma unroll
      for (int r = 0; r < 8; ++r) o[j][r] *= alpha[r];

    // P -> LDS (bf16), re-read as A fragment (two contiguous runs)
    #pragma unroll
    for (int j = 0; j < 2; ++j)
      #pragma unroll
      for (int r = 0; r < 8; ++r)
        Ps[wave][r + hi * 8][j * 16 + lanelo] = f2bf(s[j][r]);
    __syncthreads();

    ABFrag ap;
    {
      const unsigned short* row = &Ps[wave][lanelo][hi * 8];
      #pragma unroll
      for (int e = 0; e < 8; ++e) ap.u[e] = row[e];
      #pragma unroll
      for (int e = 0; e < 8; ++e) ap.u[8 + e] = row[16 + e];
    }

    // O += P * V (16x32 * 32x256); V^T fragments contiguous in key.
    #pragma unroll
    for (int jg = 0; jg < 4; ++jg) {
      ABFrag bv[4];
      #pragma unroll
      for (int j2 = 0; j2 < 4; ++j2) {
        const unsigned short* row = &VsT[(jg * 4 + j2) * 16 + lanelo][hi * 8];
        #pragma unroll
        for (int e = 0; e < 8; ++e) bv[j2].u[e] = row[e];
        #pragma unroll
        for (int e = 0; e < 8; ++e) bv[j2].u[8 + e] = row[16 + e];
      }
      #pragma unroll
      for (int j2 = 0; j2 < 4; ++j2)
        o[jg * 4 + j2] = wmma_bf16(ap, bv[j2], o[jg * 4 + j2]);
    }
    __syncthreads();   // protect Ks/VsT before restaging
  }

  // epilogue: O /= l, write enc
  float inv[8];
  #pragma unroll
  for (int r = 0; r < 8; ++r) inv[r] = 1.0f / rsum[r];
  #pragma unroll
  for (int j2 = 0; j2 < 16; ++j2)
    #pragma unroll
    for (int r = 0; r < 8; ++r) {
      int qrow = qrowBase + r + hi * 8;
      int h = j2 * 16 + lanelo;
      enc[((long)(b * Tc + qrow) * Nc + n) * Hc + h] = o[j2][r] * inv[r];
    }
}

// =====================================================================
// Launch
// =====================================================================
extern "C" void kernel_launch(void* const* d_in, const int* in_sizes, int n_in,
                              void* d_out, int out_size, void* d_ws, size_t ws_size,
                              hipStream_t stream) {
  (void)in_sizes; (void)n_in; (void)out_size; (void)ws_size;
  const float* x       = (const float*)d_in[0];
  const int*   segpos  = (const int*)d_in[1];
  /* d_in[2] attn_mask: pure causal tril; reproduced exactly by in-kernel mask */
  const float* q_w     = (const float*)d_in[3];
  const float* kv_w    = (const float*)d_in[4];
  const float* o_w     = (const float*)d_in[5];
  const float* q_scale = (const float*)d_in[6];
  const float* k_scale = (const float*)d_in[7];
  float* out = (float*)d_out;

  float* ws = (float*)d_ws;
  float* qb = ws;                                    // [B,T,N,H]
  float* kb = qb + (size_t)Bc * Tc * Nc * Hc;        // [B,T,K,H]
  float* vb = kb + (size_t)Bc * Tc * Kc * Hc;        // [B,T,K,H]
  float* eb = vb + (size_t)Bc * Tc * Kc * Hc;        // [B,T,N,H]

  const int M = Bc * Tc;                             // 4096

  // Q projection: per head n, X[M,D] * q_w[n][D,H] -> q[m, n*H + h]
  gemm_wmma<Dc, Hc, Nc * Hc, Dc><<<dim3(Hc / BN, M / BM, Nc), 256, 0, stream>>>(
      x, q_w, (long)Dc * Hc, qb, (long)Hc);
  // K projection
  gemm_wmma<Dc, Hc, Kc * Hc, Dc><<<dim3(Hc / BN, M / BM, Kc), 256, 0, stream>>>(
      x, kv_w, (long)Dc * Hc, kb, (long)Hc);
  // V projection
  gemm_wmma<Dc, Hc, Kc * Hc, Dc><<<dim3(Hc / BN, M / BM, Kc), 256, 0, stream>>>(
      x, kv_w + (size_t)Kc * Dc * Hc, (long)Dc * Hc, vb, (long)Hc);

  // RMSNorm + RoPE (q then k), one wave per head row
  normrope<<<dim3((Bc * Tc * Nc) / 8), 256, 0, stream>>>(qb, q_scale, segpos, Nc, Bc * Tc * Nc);
  normrope<<<dim3((Bc * Tc * Kc) / 8), 256, 0, stream>>>(kb, k_scale, segpos, Kc, Bc * Tc * Kc);

  // Sliding-window flash attention
  attn_wmma<<<dim3(Tc / QT, Nc, Bc), 128, 0, stream>>>(qb, kb, vb, eb);

  // Output projection: enc[M, N*H] * o_w[N*H, D] -> out[M, D]
  gemm_wmma<Nc * Hc, Dc, Dc, Nc * Hc><<<dim3(Dc / BN, M / BM, 1), 256, 0, stream>>>(
      eb, o_w, 0, out, 0);
}
